// PathwayAwareNormalization_60447369724329
// MI455X (gfx1250) — compile-verified, matrix operand order checked by reading, not run on hardware
//
#include <hip/hip_runtime.h>

// ---------------- problem constants ----------------
#define D_MODEL 1024
#define BATCH   4
#define SEQ     2048
#define NTOK    (BATCH * SEQ)      // 8192
#define NPATH   1000
#define NHEAD   8
#define HDIM    128
#define QKV_E   (3 * D_MODEL)      // 3072
#define LN_EPS  1e-5f

// ---------------- vector types ----------------
typedef __attribute__((ext_vector_type(16))) __bf16 bf16x16;
typedef __attribute__((ext_vector_type(8)))  __bf16 bf16x8;
typedef __attribute__((ext_vector_type(8)))  float  f32x8;
typedef __attribute__((ext_vector_type(4)))  unsigned int u32x4;
typedef __attribute__((ext_vector_type(8)))  int i32x8;
typedef __attribute__((ext_vector_type(4)))  int i32x4;

__device__ __forceinline__ __bf16 f2bf(float f) {
    unsigned int u = __builtin_bit_cast(unsigned int, f);
    u += 0x7FFFu + ((u >> 16) & 1u);          // round-to-nearest-even
    unsigned short h = (unsigned short)(u >> 16);
    return __builtin_bit_cast(__bf16, h);
}

__device__ __forceinline__ f32x8 zero8() {
    f32x8 z = {0.f, 0.f, 0.f, 0.f, 0.f, 0.f, 0.f, 0.f};
    return z;
}

// LDS aperture lives in addr[63:32]; low 32 bits of a flat pointer to __shared__
// are the workgroup-relative LDS byte offset required by TDM / async-LDS ops.
__device__ __forceinline__ unsigned lds_off(const void* p) {
    return (unsigned)(uintptr_t)p;
}

// A/B fragment: per lane two contiguous 16B chunks (K = 8*half..+8 and 16+8*half..+8)
__device__ __forceinline__ bf16x16 make_frag(const __bf16* p_lo, const __bf16* p_hi) {
    bf16x8 lo = *(const bf16x8*)p_lo;
    bf16x8 hi = *(const bf16x8*)p_hi;
    bf16x16 r;
#pragma unroll
    for (int i = 0; i < 8; ++i) { r[i] = lo[i]; r[i + 8] = hi[i]; }
    return r;
}

__device__ __forceinline__ f32x8 wmma_bf16(bf16x16 a, bf16x16 b, f32x8 c) {
    return __builtin_amdgcn_wmma_f32_16x16x32_bf16(false, a, false, b, (short)0, c,
                                                   false, false);
}

// ---------------- f32 -> bf16 conversion ----------------
__global__ void k_f32_to_bf16(const float* __restrict__ src, __bf16* __restrict__ dst, int n) {
    int i = blockIdx.x * blockDim.x + threadIdx.x;
    if (i < n) dst[i] = f2bf(src[i]);
}

// ---------------- pathway LayerNorm (one wave32 per token) ----------------
__global__ void __launch_bounds__(256)
k_pathway_ln(const float* __restrict__ x, const int* __restrict__ ids,
             const float* __restrict__ gamma, const float* __restrict__ beta,
             float* __restrict__ y_f32, __bf16* __restrict__ y_bf16) {
    const int wave = threadIdx.x >> 5;
    const int lane = threadIdx.x & 31;
    const int tok  = blockIdx.x * 8 + wave;
    const float* xr = x + (size_t)tok * D_MODEL;

    float s = 0.f, ss = 0.f;
    float v[32];
#pragma unroll
    for (int j = 0; j < 32; ++j) {
        float t = xr[j * 32 + lane];
        v[j] = t; s += t; ss += t * t;
    }
#pragma unroll
    for (int m = 16; m >= 1; m >>= 1) {
        s  += __shfl_xor(s,  m, 32);
        ss += __shfl_xor(ss, m, 32);
    }
    const float mu   = s * (1.f / D_MODEL);
    const float var  = ss * (1.f / D_MODEL) - mu * mu;
    const float rstd = rsqrtf(var + LN_EPS);

    const int  id    = ids[tok];
    const bool valid = (id < NPATH);
    const int  gid   = valid ? id : 0;
    const float* g  = gamma + (size_t)gid * D_MODEL;
    const float* bb = beta  + (size_t)gid * D_MODEL;
    float*  yr = y_f32  + (size_t)tok * D_MODEL;
    __bf16* yb = y_bf16 + (size_t)tok * D_MODEL;
#pragma unroll
    for (int j = 0; j < 32; ++j) {
        int e = j * 32 + lane;
        float n = (v[j] - mu) * rstd * g[e] + bb[e];
        float o = valid ? n : v[j];
        yr[e] = o;
        yb[e] = f2bf(o);
    }
}

// ---------------- generic bf16 WMMA GEMM: out = A[M,K] @ W[N,K]^T + bias (+resid) ----------------
__global__ void __launch_bounds__(256)
k_gemm_bf16(const __bf16* __restrict__ A, const __bf16* __restrict__ W,
            const float* __restrict__ bias, const float* __restrict__ resid,
            float* __restrict__ outF, __bf16* __restrict__ outB,
            int M, int N, int K) {
    const int wave = threadIdx.x >> 5;
    const int lane = threadIdx.x & 31;
    const int half = lane >> 4;
    const int l16  = lane & 15;
    const int m0 = blockIdx.x * 64  + (wave >> 2) * 32;
    const int n0 = blockIdx.y * 256 + (wave & 3) * 64;

    f32x8 acc[2][4];
#pragma unroll
    for (int i = 0; i < 2; ++i)
#pragma unroll
        for (int j = 0; j < 4; ++j) acc[i][j] = zero8();

    for (int kk = 0; kk < K; kk += 32) {
        if (kk + 32 < K) {
            __builtin_prefetch(A + (size_t)(m0 + l16) * K + kk + 32, 0, 3);
            __builtin_prefetch(W + (size_t)(n0 + l16) * K + kk + 32, 0, 3);
        }
        bf16x16 af[2], bfr[4];
#pragma unroll
        for (int i = 0; i < 2; ++i) {
            const __bf16* base = A + (size_t)(m0 + 16 * i + l16) * K + kk + 8 * half;
            af[i] = make_frag(base, base + 16);
        }
#pragma unroll
        for (int j = 0; j < 4; ++j) {
            const __bf16* base = W + (size_t)(n0 + 16 * j + l16) * K + kk + 8 * half;
            bfr[j] = make_frag(base, base + 16);
        }
#pragma unroll
        for (int i = 0; i < 2; ++i)
#pragma unroll
            for (int j = 0; j < 4; ++j)
                acc[i][j] = wmma_bf16(af[i], bfr[j], acc[i][j]);
    }

#pragma unroll
    for (int i = 0; i < 2; ++i) {
#pragma unroll
        for (int j = 0; j < 4; ++j) {
            const int col = n0 + 16 * j + l16;
            const float bv = bias[col];
#pragma unroll
            for (int r = 0; r < 8; ++r) {
                const int row = m0 + 16 * i + r + 8 * half;
                float val = acc[i][j][r] + bv;
                if (resid) val += resid[(size_t)row * N + col];
                if (outF)  outF[(size_t)row * N + col] = val;
                if (outB)  outB[(size_t)row * N + col] = f2bf(val);
            }
        }
    }
}

// ---------------- flash attention ----------------
// 4 waves x 16 q-rows per block; 32-key tiles.
// K tile staged by the Tensor Data Mover (tensor_load_to_lds, TENSORcnt);
// V tile staged by per-lane async copies (global_load_async_to_lds_b128, ASYNCcnt),
// then transposed in LDS for the P x V B-fragments.
__global__ void __launch_bounds__(128)
k_attention(const __bf16* __restrict__ qkv,   // [NTOK][3072] : q|k|v head-major
            __bf16* __restrict__ ctx) {       // [NTOK][1024]
    __shared__ __bf16 ldsK[32 * HDIM];        // K tile row-major [key][dim], 8 KB (TDM dest)
    __shared__ __bf16 ldsVrow[32 * HDIM];     // V tile row-major [key][dim], 8 KB (async dest)
    __shared__ __bf16 ldsVt[HDIM * 32];       // V^T tile [dim][key], 8 KB
    __shared__ __bf16 ldsP[4][16 * 32];       // per-wave P transpose, 4 KB

    const int wave = threadIdx.x >> 5;
    const int lane = threadIdx.x & 31;
    const int half = lane >> 4;
    const int l16  = lane & 15;
    const int b = blockIdx.y >> 3;
    const int h = blockIdx.y & 7;
    const int q0 = blockIdx.x * 64 + wave * 16;
    const float scale = 0.08838834764831845f;  // 1/sqrt(128)

    // Q fragments stay resident in VGPRs for the whole key loop
    bf16x16 qf[4];
#pragma unroll
    for (int t = 0; t < 4; ++t) {
        const __bf16* base =
            qkv + (size_t)(b * SEQ + q0 + l16) * QKV_E + h * HDIM + 32 * t + 8 * half;
        qf[t] = make_frag(base, base + 16);
    }

    f32x8 cacc[8];
#pragma unroll
    for (int g = 0; g < 8; ++g) cacc[g] = zero8();
    float mst[8], lst[8];
#pragma unroll
    for (int r = 0; r < 8; ++r) { mst[r] = -1e30f; lst[r] = 0.f; }

    const __bf16* kbase = qkv + (size_t)b * SEQ * QKV_E + D_MODEL + h * HDIM;
    const __bf16* vbase = qkv + (size_t)b * SEQ * QKV_E + 2 * D_MODEL + h * HDIM;

    for (int kt = 0; kt < SEQ; kt += 32) {
        __syncthreads();  // previous tile's LDS consumers done

        // ---- TDM: K tile [32 x 128] bf16 -> ldsK, one descriptor, wave 0 issues ----
        if (wave == 0) {
            unsigned long long ga =
                (unsigned long long)(uintptr_t)(kbase + (size_t)kt * QKV_E);
            unsigned la = lds_off(&ldsK[0]);
            u32x4 g0;
            g0.x = 1u;  // count=1, is_restore=0, gather off
            g0.y = (unsigned)__builtin_amdgcn_readfirstlane((int)la);
            g0.z = (unsigned)__builtin_amdgcn_readfirstlane((int)(unsigned)(ga & 0xFFFFFFFFull));
            g0.w = ((unsigned)__builtin_amdgcn_readfirstlane(
                        (int)(unsigned)((ga >> 32) & 0x01FFFFFFull))) | (2u << 30); // type=2
            i32x8 g1;
            g1[0] = (int)(1u << 16);      // data_size = 1 -> 2 bytes
            g1[1] = (int)(128u << 16);    // tensor_dim0 = 128 (lo16 at [31:16])
            g1[2] = (int)(2048u << 16);   // tensor_dim0 hi=0, tensor_dim1 lo = 2048
            g1[3] = (int)(128u << 16);    // tensor_dim1 hi=0, tile_dim0 = 128
            g1[4] = 32;                   // tile_dim1 = 32, tile_dim2 = 0
            g1[5] = 3072;                 // tensor_dim0_stride = 3072 elems
            g1[6] = 0;
            g1[7] = 0;
            i32x4 gz = {0, 0, 0, 0};
#if defined(__clang_major__) && __clang_major__ >= 23
            i32x8 gp8 = {0, 0, 0, 0, 0, 0, 0, 0};
            __builtin_amdgcn_tensor_load_to_lds(g0, g1, gz, gz, gp8, 0);
#else
            __builtin_amdgcn_tensor_load_to_lds(g0, g1, gz, gz, 0);
#endif
            __builtin_amdgcn_s_wait_tensorcnt(0);
        }

        // ---- async: V tile [32 x 128] bf16 -> ldsVrow (per-lane b128 copies) ----
        {
            const int tid = threadIdx.x;
#pragma unroll
            for (int i = 0; i < 4; ++i) {
                const int chunk = tid + 128 * i;   // 512 x 16B = 8 KB
                const int key   = chunk >> 4;
                const int part  = chunk & 15;
                const __bf16* gp = vbase + (size_t)(kt + key) * QKV_E + part * 8;
                unsigned la = lds_off(&ldsVrow[key * HDIM + part * 8]);
                asm volatile("global_load_async_to_lds_b128 %0, %1, off"
                             :: "v"(la), "v"(gp) : "memory");
            }
            asm volatile("s_wait_asynccnt 0x0" ::: "memory");
        }
        __syncthreads();  // ldsK (TDM) + ldsVrow (async) ready

        // ---- transpose V in LDS: ldsVrow[key][dim] -> ldsVt[dim][key] ----
        {
            const int key = threadIdx.x >> 2;
            const int d0  = (threadIdx.x & 3) * 32;
#pragma unroll
            for (int c = 0; c < 4; ++c) {
                bf16x8 vv = *(const bf16x8*)&ldsVrow[key * HDIM + d0 + 8 * c];
#pragma unroll
                for (int e = 0; e < 8; ++e) ldsVt[(d0 + 8 * c + e) * 32 + key] = vv[e];
            }
        }

        // ---- scores: S(16x32) = Q(16x128) x K^T, K frags from LDS (shared by all waves) ----
        f32x8 sacc[2] = {zero8(), zero8()};
#pragma unroll
        for (int jj = 0; jj < 2; ++jj) {
#pragma unroll
            for (int t = 0; t < 4; ++t) {
                const __bf16* base = &ldsK[(16 * jj + l16) * HDIM + 32 * t + 8 * half];
                bf16x16 kf = make_frag(base, base + 16);
                sacc[jj] = wmma_bf16(qf[t], kf, sacc[jj]);
            }
        }

        // ---- online softmax update + stage P for transpose ----
#pragma unroll
        for (int r = 0; r < 8; ++r) {
            float s0 = sacc[0][r] * scale;
            float s1 = sacc[1][r] * scale;
            float mx = fmaxf(s0, s1);
#pragma unroll
            for (int m = 8; m >= 1; m >>= 1) mx = fmaxf(mx, __shfl_xor(mx, m, 16));
            float mnew  = fmaxf(mst[r], mx);
            float alpha = __expf(mst[r] - mnew);
            float p0 = __expf(s0 - mnew);
            float p1 = __expf(s1 - mnew);
            float rs = p0 + p1;
#pragma unroll
            for (int m = 8; m >= 1; m >>= 1) rs += __shfl_xor(rs, m, 16);
            lst[r] = lst[r] * alpha + rs;
            mst[r] = mnew;
#pragma unroll
            for (int g = 0; g < 8; ++g) cacc[g][r] *= alpha;
            const int prow = r + 8 * half;
            ldsP[wave][prow * 32 + l16]      = f2bf(p0);
            ldsP[wave][prow * 32 + 16 + l16] = f2bf(p1);
        }
        __syncthreads();  // ldsVt + ldsP visible

        // ---- ctx(16x128) += P(16x32) x V(32x128) ----
        bf16x16 pf = make_frag(&ldsP[wave][l16 * 32 + 8 * half],
                               &ldsP[wave][l16 * 32 + 16 + 8 * half]);
#pragma unroll
        for (int g = 0; g < 8; ++g) {
            const __bf16* base = &ldsVt[(16 * g + l16) * 32 + 8 * half];
            bf16x16 vf = make_frag(base, base + 16);
            cacc[g] = wmma_bf16(pf, vf, cacc[g]);
        }
    }

    // epilogue: normalize by l and emit ctx
#pragma unroll
    for (int r = 0; r < 8; ++r) {
        const float inv = 1.f / lst[r];
        const int tok = b * SEQ + q0 + r + 8 * half;
#pragma unroll
        for (int g = 0; g < 8; ++g)
            ctx[(size_t)tok * D_MODEL + h * HDIM + 16 * g + l16] = f2bf(cacc[g][r] * inv);
    }
}

// ---------------- host orchestration ----------------
extern "C" void kernel_launch(void* const* d_in, const int* in_sizes, int n_in,
                              void* d_out, int out_size, void* d_ws, size_t ws_size,
                              hipStream_t stream) {
    const float* x       = (const float*)d_in[0];
    const int*   ids     = (const int*)  d_in[1];
    const float* gamma   = (const float*)d_in[2];
    const float* beta    = (const float*)d_in[3];
    const float* w_inter = (const float*)d_in[4];
    const float* b_inter = (const float*)d_in[5];
    const float* w_qkv   = (const float*)d_in[6];
    const float* b_qkv   = (const float*)d_in[7];
    const float* w_out   = (const float*)d_in[8];
    const float* b_out   = (const float*)d_in[9];
    float* out = (float*)d_out;

    char* ws = (char*)d_ws;
    size_t off = 0;
    auto alloc = [&](size_t bytes) -> char* {
        char* p = ws + off;
        off += (bytes + 255) & ~(size_t)255;
        return p;
    };
    float*  x_ln_f = (float*) alloc((size_t)NTOK * D_MODEL * 4);
    __bf16* x_ln_b = (__bf16*)alloc((size_t)NTOK * D_MODEL * 2);
    float*  x1_f   = (float*) alloc((size_t)NTOK * D_MODEL * 4);
    __bf16* x1_b   = (__bf16*)alloc((size_t)NTOK * D_MODEL * 2);
    __bf16* qkv_b  = (__bf16*)alloc((size_t)NTOK * QKV_E * 2);
    __bf16* ctx_b  = (__bf16*)alloc((size_t)NTOK * D_MODEL * 2);
    __bf16* wi_b   = (__bf16*)alloc((size_t)D_MODEL * D_MODEL * 2);
    __bf16* wq_b   = (__bf16*)alloc((size_t)QKV_E * D_MODEL * 2);
    __bf16* wo_b   = (__bf16*)alloc((size_t)D_MODEL * D_MODEL * 2);

    k_f32_to_bf16<<<(D_MODEL * D_MODEL + 255) / 256, 256, 0, stream>>>(w_inter, wi_b, D_MODEL * D_MODEL);
    k_f32_to_bf16<<<(QKV_E * D_MODEL + 255) / 256, 256, 0, stream>>>(w_qkv, wq_b, QKV_E * D_MODEL);
    k_f32_to_bf16<<<(D_MODEL * D_MODEL + 255) / 256, 256, 0, stream>>>(w_out, wo_b, D_MODEL * D_MODEL);

    k_pathway_ln<<<NTOK / 8, 256, 0, stream>>>(x, ids, gamma, beta, x_ln_f, x_ln_b);

    dim3 g1(NTOK / 64, D_MODEL / 256);
    k_gemm_bf16<<<g1, 256, 0, stream>>>(x_ln_b, wi_b, b_inter, x_ln_f, x1_f, x1_b,
                                        NTOK, D_MODEL, D_MODEL);

    dim3 g2(NTOK / 64, QKV_E / 256);
    k_gemm_bf16<<<g2, 256, 0, stream>>>(x1_b, wq_b, b_qkv, nullptr, nullptr, qkv_b,
                                        NTOK, QKV_E, D_MODEL);

    dim3 g3(SEQ / 64, BATCH * NHEAD);
    k_attention<<<g3, 128, 0, stream>>>(qkv_b, ctx_b);

    dim3 g4(NTOK / 64, D_MODEL / 256);
    k_gemm_bf16<<<g4, 256, 0, stream>>>(ctx_b, wo_b, b_out, x1_f, out, nullptr,
                                        NTOK, D_MODEL, D_MODEL);
}
